// MoELayer_712964571352
// MI455X (gfx1250) — matrix operand order repaired and checked by reading, not compiled, not run on hardware
//
#include <hip/hip_runtime.h>
#include <math.h>

// ---------------------------------------------------------------------------
// MoE layer for MI455X (gfx1250): bf16 WMMA 16x16x32, f32 accumulate.
// Sizes fixed by the reference: BT=4096 tokens, D=1024, H=4096, E=8, TOPK=2.
// ---------------------------------------------------------------------------

#define BT    4096
#define DDIM  1024
#define HDIM  4096
#define NEXP  8
#define TILE_K 32
#define LDS_STRIDE 20   // u32 per LDS row: 16 data + 4 pad; 16B-aligned rows,
                        // banks 20*i mod 64 are conflict-free for 16-lane reads

typedef __bf16 bf16_t;
typedef bf16_t v16bf __attribute__((ext_vector_type(16)));
typedef float  v8f   __attribute__((ext_vector_type(8)));

union FragU { v16bf v; unsigned int u[8]; uint4 q[2]; };

__device__ __forceinline__ unsigned short f32_to_bf16(float f) {
    unsigned int u = __float_as_uint(f);
    u += 0x7FFFu + ((u >> 16) & 1u);           // round-to-nearest-even
    return (unsigned short)(u >> 16);
}
__device__ __forceinline__ unsigned int pack2_bf16(float lo, float hi) {
    return (unsigned int)f32_to_bf16(lo) | ((unsigned int)f32_to_bf16(hi) << 16);
}

// ---------------------------------------------------------------------------
// Router: logits = x @ wg, softmax, top-2.  One wave (32 lanes) per token.
// ctrl layout (ints): [0..7]=counts  [8..15]=cursors  [16..24]=offsets
// ---------------------------------------------------------------------------
__global__ void __launch_bounds__(256)
moe_router(const float* __restrict__ x, const float* __restrict__ wg,
           float* __restrict__ logits_out, int* __restrict__ topi,
           float* __restrict__ wts, int* __restrict__ ctrl)
{
    const int tid  = threadIdx.x;
    const int lane = tid & 31;
    const int wid  = tid >> 5;
    const int t = blockIdx.x * 8 + wid;
    if (t >= BT) return;

    const float* xr = x + (size_t)t * DDIM;
    float acc[NEXP];
#pragma unroll
    for (int e = 0; e < NEXP; ++e) acc[e] = 0.0f;

    for (int d = lane; d < DDIM; d += 32) {
        const float xv = xr[d];
        const float* wr = wg + (size_t)d * NEXP;
#pragma unroll
        for (int e = 0; e < NEXP; ++e) acc[e] = fmaf(xv, wr[e], acc[e]);
    }
#pragma unroll
    for (int off = 16; off > 0; off >>= 1) {
#pragma unroll
        for (int e = 0; e < NEXP; ++e) acc[e] += __shfl_down(acc[e], off, 32);
    }

    if (lane == 0) {
        float mx = acc[0];
#pragma unroll
        for (int e = 1; e < NEXP; ++e) mx = fmaxf(mx, acc[e]);
        float p[NEXP], s = 0.0f;
#pragma unroll
        for (int e = 0; e < NEXP; ++e) { p[e] = expf(acc[e] - mx); s += p[e]; }
        const float inv = 1.0f / s;
#pragma unroll
        for (int e = 0; e < NEXP; ++e) p[e] *= inv;

        // top-2 (tie -> lowest index, matching jax.lax.top_k)
        int i0 = 0;
#pragma unroll
        for (int e = 1; e < NEXP; ++e) if (p[e] > p[i0]) i0 = e;
        int i1 = (i0 == 0) ? 1 : 0;
#pragma unroll
        for (int e = 0; e < NEXP; ++e) if (e != i0 && p[e] > p[i1]) i1 = e;

#pragma unroll
        for (int e = 0; e < NEXP; ++e) logits_out[(size_t)t * NEXP + e] = acc[e];
        topi[t * 2 + 0] = i0;
        topi[t * 2 + 1] = i1;
        // Bug-faithful to the reference: weight for slot j is probs[:, j]
        // (softmax prob of expert COLUMN j), not the top-k weight.
        wts[t * 2 + 0] = p[0];
        wts[t * 2 + 1] = p[1];
        atomicAdd(&ctrl[i0], 1);
        atomicAdd(&ctrl[i1], 1);
    }
}

__global__ void moe_scan(int* ctrl)
{
    if (threadIdx.x == 0) {
        int s = 0;
        for (int e = 0; e < NEXP; ++e) { ctrl[16 + e] = s; s += ctrl[e]; }
        ctrl[24] = s;   // == 2*BT
    }
}

__global__ void __launch_bounds__(256)
moe_place(const int* __restrict__ topi, const float* __restrict__ wts,
          int* ctrl, int* __restrict__ rows, float* __restrict__ rwt)
{
    const int t = blockIdx.x * blockDim.x + threadIdx.x;
    if (t >= BT) return;
#pragma unroll
    for (int j = 0; j < 2; ++j) {
        const int e   = topi[t * 2 + j];
        const int pos = atomicAdd(&ctrl[8 + e], 1);
        const int row = ctrl[16 + e] + pos;
        rows[row] = t;
        rwt[row]  = wts[t * 2 + j];
    }
}

// ---------------------------------------------------------------------------
// GEMM1: h[row, :] = gelu( x[tok(row), :] @ w1[e] + b1[e] ),  bf16 out.
// Block tile 128(M) x 128(N), K-step 32.  8 waves arranged 2(M) x 4(N),
// each wave owns 64x32 = 4x2 WMMA 16x16 tiles.
// ---------------------------------------------------------------------------
__global__ void __launch_bounds__(256)
moe_gemm1(const float* __restrict__ x, const float* __restrict__ w1,
          const float* __restrict__ b1, const int* __restrict__ rows,
          const int* __restrict__ ctrl, unsigned short* __restrict__ hbuf)
{
    __shared__ unsigned int ldsA[128 * LDS_STRIDE];
    __shared__ unsigned int ldsB[128 * LDS_STRIDE];

    const int e        = blockIdx.z;
    const int rowStart = ctrl[16 + e] + blockIdx.y * 128;
    const int rowEnd   = ctrl[16 + e + 1];
    if (rowStart >= rowEnd) return;
    const int n0 = blockIdx.x * 128;

    const float* w1e = w1 + (size_t)e * DDIM * HDIM;
    const float* b1e = b1 + (size_t)e * HDIM;

    const int tid     = threadIdx.x;
    const int lane    = tid & 31;
    const int wid     = tid >> 5;
    const int wm      = wid >> 2;     // 0..1
    const int wn      = wid & 3;      // 0..3
    const int halfSel = lane >> 4;    // 0/1
    const int lmod    = lane & 15;

    // A-fill: thread -> (row, 16-element k segment); clamped row is always a
    // valid gather entry so loads are unconditional (no exec divergence).
    const int  ar     = tid >> 1;
    const int  aks    = (tid & 1) * 16;        // k base in halves
    const int  agrow  = rowStart + ar;
    const bool avalid = agrow < rowEnd;
    const int  atok   = rows[avalid ? agrow : (rowEnd - 1)];
    const float* axp  = x + (size_t)atok * DDIM + aks;
    unsigned int* aSt = &ldsA[ar * LDS_STRIDE + (aks >> 1)];

    // B-fill: thread -> (column n, 16-row k segment)
    const int bn  = tid & 127;
    const int bkg = (tid >> 7) * 16;
    unsigned int* bSt = &ldsB[bn * LDS_STRIDE + (bkg >> 1)];

    v8f acc[4][2] = {};

    for (int k0 = 0; k0 < DDIM; k0 += TILE_K) {
        // stage A (gathered token rows, f32 -> bf16), 4x b128 loads
        float4 f0 = *(const float4*)(axp + k0);
        float4 f1 = *(const float4*)(axp + k0 + 4);
        float4 f2 = *(const float4*)(axp + k0 + 8);
        float4 f3 = *(const float4*)(axp + k0 + 12);
        if (!avalid) {
            f0 = make_float4(0.f, 0.f, 0.f, 0.f);
            f1 = f0; f2 = f0; f3 = f0;
        }
        ((uint4*)aSt)[0] = make_uint4(pack2_bf16(f0.x, f0.y), pack2_bf16(f0.z, f0.w),
                                      pack2_bf16(f1.x, f1.y), pack2_bf16(f1.z, f1.w));
        ((uint4*)aSt)[1] = make_uint4(pack2_bf16(f2.x, f2.y), pack2_bf16(f2.z, f2.w),
                                      pack2_bf16(f3.x, f3.y), pack2_bf16(f3.z, f3.w));

        // stage B transposed: ldsB[n][k]  (w1 is [K][N] row-major, N coalesced)
        const float* bp = w1e + (size_t)(k0 + bkg) * HDIM + n0 + bn;
        unsigned int bu[8];
#pragma unroll
        for (int j = 0; j < 8; ++j)
            bu[j] = pack2_bf16(bp[(size_t)(2 * j) * HDIM], bp[(size_t)(2 * j + 1) * HDIM]);
        ((uint4*)bSt)[0] = make_uint4(bu[0], bu[1], bu[2], bu[3]);
        ((uint4*)bSt)[1] = make_uint4(bu[4], bu[5], bu[6], bu[7]);
        __syncthreads();

        FragU aF[4], bF[2];
#pragma unroll
        for (int m = 0; m < 4; ++m) {
            const uint4* base =
                (const uint4*)&ldsA[(wm * 64 + m * 16 + lmod) * LDS_STRIDE + halfSel * 4];
            aF[m].q[0] = base[0];          // u[0..3]
            aF[m].q[1] = base[2];          // u[4..7] at +8 dwords
        }
#pragma unroll
        for (int n = 0; n < 2; ++n) {
            const uint4* base =
                (const uint4*)&ldsB[(wn * 32 + n * 16 + lmod) * LDS_STRIDE + halfSel * 4];
            bF[n].q[0] = base[0];
            bF[n].q[1] = base[2];
        }
#pragma unroll
        for (int m = 0; m < 4; ++m)
#pragma unroll
            for (int n = 0; n < 2; ++n)
                acc[m][n] = __builtin_amdgcn_wmma_f32_16x16x32_bf16(
                    false, aF[m].v, false, bF[n].v, (short)0, acc[m][n], false, false);
        __syncthreads();
    }

    // epilogue: bias + exact GELU, store bf16 h
#pragma unroll
    for (int n = 0; n < 2; ++n) {
        const int   gcol = n0 + wn * 32 + n * 16 + lmod;
        const float bias = b1e[gcol];
#pragma unroll
        for (int m = 0; m < 4; ++m) {
            const int rbase = wm * 64 + m * 16 + halfSel * 8;
#pragma unroll
            for (int r = 0; r < 8; ++r) {
                const int grow = rowStart + rbase + r;
                if (grow < rowEnd) {
                    float v = acc[m][n][r] + bias;
                    v = 0.5f * v * (1.0f + erff(v * 0.70710678118654752f));
                    hbuf[(size_t)grow * HDIM + gcol] = f32_to_bf16(v);
                }
            }
        }
    }
}

// ---------------------------------------------------------------------------
// GEMM2: y[row, :] = h[row, :] @ w2[e] + b2[e];  out[tok(row)] += wt(row)*y
// (exactly 2 commutative f32 atomic adds per output element -> deterministic)
// ---------------------------------------------------------------------------
__global__ void __launch_bounds__(256)
moe_gemm2(const unsigned short* __restrict__ hbuf, const float* __restrict__ w2,
          const float* __restrict__ b2, const int* __restrict__ rows,
          const float* __restrict__ rwt, const int* __restrict__ ctrl,
          float* __restrict__ out)
{
    __shared__ unsigned int ldsA[128 * LDS_STRIDE];
    __shared__ unsigned int ldsB[128 * LDS_STRIDE];

    const int e        = blockIdx.z;
    const int rowStart = ctrl[16 + e] + blockIdx.y * 128;
    const int rowEnd   = ctrl[16 + e + 1];
    if (rowStart >= rowEnd) return;
    const int n0 = blockIdx.x * 128;

    const float* w2e = w2 + (size_t)e * HDIM * DDIM;
    const float* b2e = b2 + (size_t)e * DDIM;

    const int tid     = threadIdx.x;
    const int lane    = tid & 31;
    const int wid     = tid >> 5;
    const int wm      = wid >> 2;
    const int wn      = wid & 3;
    const int halfSel = lane >> 4;
    const int lmod    = lane & 15;

    const int  ar     = tid >> 1;
    const int  aks    = (tid & 1) * 16;
    const int  agrow  = rowStart + ar;
    const bool avalid = agrow < rowEnd;
    const int  arowc  = avalid ? agrow : (rowEnd - 1);
    const uint4* ahq  = (const uint4*)(hbuf + (size_t)arowc * HDIM + aks);
    unsigned int* aSt = &ldsA[ar * LDS_STRIDE + (aks >> 1)];

    const int bn  = tid & 127;
    const int bkg = (tid >> 7) * 16;
    unsigned int* bSt = &ldsB[bn * LDS_STRIDE + (bkg >> 1)];

    v8f acc[4][2] = {};

    for (int k0 = 0; k0 < HDIM; k0 += TILE_K) {
        // stage A: already bf16, 2x b128 copies (zero for pad rows)
        uint4 h0 = ahq[(k0 >> 3) + 0];
        uint4 h1 = ahq[(k0 >> 3) + 1];
        if (!avalid) { h0 = make_uint4(0u, 0u, 0u, 0u); h1 = h0; }
        ((uint4*)aSt)[0] = h0;
        ((uint4*)aSt)[1] = h1;

        // stage B transposed (w2 is [K=H][N=D] row-major)
        const float* bp = w2e + (size_t)(k0 + bkg) * DDIM + n0 + bn;
        unsigned int bu[8];
#pragma unroll
        for (int j = 0; j < 8; ++j)
            bu[j] = pack2_bf16(bp[(size_t)(2 * j) * DDIM], bp[(size_t)(2 * j + 1) * DDIM]);
        ((uint4*)bSt)[0] = make_uint4(bu[0], bu[1], bu[2], bu[3]);
        ((uint4*)bSt)[1] = make_uint4(bu[4], bu[5], bu[6], bu[7]);
        __syncthreads();

        FragU aF[4], bF[2];
#pragma unroll
        for (int m = 0; m < 4; ++m) {
            const uint4* base =
                (const uint4*)&ldsA[(wm * 64 + m * 16 + lmod) * LDS_STRIDE + halfSel * 4];
            aF[m].q[0] = base[0];
            aF[m].q[1] = base[2];
        }
#pragma unroll
        for (int n = 0; n < 2; ++n) {
            const uint4* base =
                (const uint4*)&ldsB[(wn * 32 + n * 16 + lmod) * LDS_STRIDE + halfSel * 4];
            bF[n].q[0] = base[0];
            bF[n].q[1] = base[2];
        }
#pragma unroll
        for (int m = 0; m < 4; ++m)
#pragma unroll
            for (int n = 0; n < 2; ++n)
                acc[m][n] = __builtin_amdgcn_wmma_f32_16x16x32_bf16(
                    false, aF[m].v, false, bF[n].v, (short)0, acc[m][n], false, false);
        __syncthreads();
    }

    // epilogue: bias, slot-weight scale, scatter-add to final output
    const int   gcol0 = n0 + wn * 32 + lmod;
    const float bias0 = b2e[gcol0];
    const float bias1 = b2e[gcol0 + 16];
#pragma unroll
    for (int m = 0; m < 4; ++m) {
        const int rbase = wm * 64 + m * 16 + halfSel * 8;
#pragma unroll
        for (int r = 0; r < 8; ++r) {
            const int grow = rowStart + rbase + r;
            if (grow < rowEnd) {
                const int   tok = rows[grow];
                const float wt  = rwt[grow];
                float* orow = out + (size_t)tok * DDIM;
                atomicAdd(orow + gcol0,      wt * (acc[m][0][r] + bias0));
                atomicAdd(orow + gcol0 + 16, wt * (acc[m][1][r] + bias1));
            }
        }
    }
}

// ---------------------------------------------------------------------------
// Launch.  d_in order: x, wg, w1, b1, w2, b2.  d_out: final (BT*D) ++ logits.
// ws: [ctrl 1KB][topi 32KB][wts 32KB][rows 32KB][rwt 32KB][hbuf 64MB]
// ---------------------------------------------------------------------------
extern "C" void kernel_launch(void* const* d_in, const int* in_sizes, int n_in,
                              void* d_out, int out_size, void* d_ws, size_t ws_size,
                              hipStream_t stream)
{
    const float* x  = (const float*)d_in[0];
    const float* wg = (const float*)d_in[1];
    const float* w1 = (const float*)d_in[2];
    const float* b1 = (const float*)d_in[3];
    const float* w2 = (const float*)d_in[4];
    const float* b2 = (const float*)d_in[5];

    float* out    = (float*)d_out;
    float* logits = out + (size_t)BT * DDIM;

    char*  ws   = (char*)d_ws;
    int*   ctrl = (int*)ws;                                    // 1024 B
    int*   topi = (int*)(ws + 1024);                           // 32 KB
    float* wts  = (float*)(ws + 1024 + 32768);                 // 32 KB
    int*   rows = (int*)(ws + 1024 + 2 * 32768);               // 32 KB
    float* rwt  = (float*)(ws + 1024 + 3 * 32768);             // 32 KB
    unsigned short* hbuf = (unsigned short*)(ws + 1024 + 4 * 32768); // 64 MB bf16

    hipMemsetAsync(out, 0, (size_t)BT * DDIM * sizeof(float), stream);
    hipMemsetAsync(ctrl, 0, 1024, stream);

    moe_router<<<BT / 8, 256, 0, stream>>>(x, wg, logits, topi, wts, ctrl);
    moe_scan<<<1, 32, 0, stream>>>(ctrl);
    moe_place<<<BT / 256, 256, 0, stream>>>(topi, wts, ctrl, rows, rwt);
    moe_gemm1<<<dim3(HDIM / 128, 32, NEXP), 256, 0, stream>>>(x, w1, b1, rows, ctrl, hbuf);
    moe_gemm2<<<dim3(DDIM / 128, 32, NEXP), 256, 0, stream>>>(hbuf, w2, b2, rows, rwt, ctrl, out);
}
